// DrumTransformer_59923383713874
// MI455X (gfx1250) — matrix operand order repaired and checked by reading, not compiled, not run on hardware
//
#include <hip/hip_runtime.h>
#include <hip/hip_bf16.h>

// ---------------- model constants ----------------
#define LAYERS 6
#define BB 8
#define TT 1024
#define CC 768
#define HH 12
#define HD 64
#define DFF 3072
#define VV 512
#define MTOK (BB*TT)          // 8192 tokens
#define NEG_BIG -1.0e30f

typedef __attribute__((ext_vector_type(16))) __bf16 v16bf;
typedef __attribute__((ext_vector_type(8)))  float  v8f;

// ---------------- WMMA helpers ----------------
__device__ __forceinline__ v8f wmma_bf16(v16bf a, v16bf b, v8f c) {
  return __builtin_amdgcn_wmma_f32_16x16x32_bf16(false, a, false, b, (short)0, c, false, false);
}

// A fragment (16x32 bf16, row-major source with leading dim `ld`):
// lane<16: row=lane, K = {0..7, 16..23}; lane>=16: row=lane-16, K = {8..15, 24..31}
__device__ __forceinline__ v16bf load_fragA(const __bf16* base, int ld, int lane) {
  union { v16bf v; uint4 q[2]; } f;
  int r = lane & 15, kb = (lane >> 4) << 3;
  f.q[0] = *(const uint4*)(base + (size_t)r * ld + kb);
  f.q[1] = *(const uint4*)(base + (size_t)r * ld + kb + 16);
  return f.v;
}

// B fragment (32x16 bf16) loaded from Bt = B-transposed [N][K] rows with leading dim `ld`:
// lane<16: col N=lane, K = {0..7, 16..23}; lane>=16: col N=lane-16, K = {8..15, 24..31}
__device__ __forceinline__ v16bf load_fragB(const __bf16* base, int ld, int lane) {
  union { v16bf v; uint4 q[2]; } f;
  int r = lane & 15, kb = (lane >> 4) << 3;
  f.q[0] = *(const uint4*)(base + (size_t)r * ld + kb);
  f.q[1] = *(const uint4*)(base + (size_t)r * ld + kb + 16);
  return f.v;
}

// CDNA5 async global->LDS copy, 16 bytes per lane (GVS mode: SGPR base + VGPR byte offset)
__device__ __forceinline__ void async_copy16(unsigned int lds_addr, const void* sbase,
                                             unsigned int voff) {
  asm volatile("global_load_async_to_lds_b128 %0, %1, %2"
               :: "v"(lds_addr), "v"(voff), "s"(sbase)
               : "memory");
}
__device__ __forceinline__ void wait_async0() {
  asm volatile("s_wait_asynccnt 0x0" ::: "memory");
}

__device__ __forceinline__ float gelu_exact(float v) {
  return 0.5f * v * (1.0f + erff(v * 0.7071067811865476f));
}

// ---------------- elementwise kernels ----------------
__global__ void embed_kernel(const int* __restrict__ idx, const float* __restrict__ tok,
                             const float* __restrict__ pos, float* __restrict__ x) {
  int id = blockIdx.x * 256 + threadIdx.x;           // over MTOK*CC
  if (id >= MTOK * CC) return;
  int bt = id / CC, c = id - bt * CC;
  int t = bt & (TT - 1);
  x[id] = tok[(size_t)idx[bt] * CC + c] + pos[(size_t)t * CC + c];
}

// src f32 [K,N] -> dst bf16 [N,K]  (weight transpose+convert)
__global__ void wt_transpose_kernel(const float* __restrict__ src, __bf16* __restrict__ dst,
                                    int K, int N) {
  int id = blockIdx.x * 256 + threadIdx.x;
  if (id >= K * N) return;
  int n = id / K, k = id - n * K;
  dst[id] = (__bf16)src[(size_t)k * N + n];
}

__global__ void convert_kernel(const float* __restrict__ src, __bf16* __restrict__ dst, int n) {
  int id = blockIdx.x * 256 + threadIdx.x;
  if (id < n) dst[id] = (__bf16)src[id];
}

// layernorm over CC=768, block per row, bf16 output
__global__ __launch_bounds__(256) void layernorm_kernel(const float* __restrict__ x,
    const float* __restrict__ g, const float* __restrict__ b, __bf16* __restrict__ out) {
  __shared__ float red[256];
  int row = blockIdx.x, tid = threadIdx.x;
  const float* xr = x + (size_t)row * CC;
  float v[3];
#pragma unroll
  for (int i = 0; i < 3; i++) v[i] = xr[tid + i * 256];
  red[tid] = v[0] + v[1] + v[2];
  __syncthreads();
  for (int st = 128; st > 0; st >>= 1) { if (tid < st) red[tid] += red[tid + st]; __syncthreads(); }
  float mu = red[0] * (1.0f / CC);
  __syncthreads();
  float d0 = v[0] - mu, d1 = v[1] - mu, d2 = v[2] - mu;
  red[tid] = d0 * d0 + d1 * d1 + d2 * d2;
  __syncthreads();
  for (int st = 128; st > 0; st >>= 1) { if (tid < st) red[tid] += red[tid + st]; __syncthreads(); }
  float rs = rsqrtf(red[0] * (1.0f / CC) + 1e-5f);
#pragma unroll
  for (int i = 0; i < 3; i++) {
    int c = tid + i * 256;
    out[(size_t)row * CC + c] = (__bf16)((v[i] - mu) * rs * g[c] + b[c]);
  }
}

// qkv bf16 [MTOK, 3C] -> q/k/v bf16 [B,H,T,HD]
__global__ void qkv_split_kernel(const __bf16* __restrict__ qkv, __bf16* __restrict__ q,
                                 __bf16* __restrict__ k, __bf16* __restrict__ v) {
  int id = blockIdx.x * 256 + threadIdx.x;           // over MTOK*CC
  if (id >= MTOK * CC) return;
  int bt = id / CC, c = id - bt * CC;
  int b = bt / TT, t = bt - b * TT;
  int h = c / HD, hd = c - h * HD;
  size_t dst = (((size_t)(b * HH + h)) * TT + t) * HD + hd;
  size_t src = (size_t)bt * (3 * CC) + c;
  q[dst] = qkv[src];
  k[dst] = qkv[src + CC];
  v[dst] = qkv[src + 2 * CC];
}

// ---------------- WMMA GEMM (async double-buffered) ----------------
// Y[M,N] = act(X[M,K](bf16) * Wt[N,K](bf16)^T + bias) (+ residual)
// block tile 64(M) x 128(N), BK=32, 256 threads = 8 waves, wave tile 32x32 (4 wmma / k-step)
// Global->LDS moves via GLOBAL_LOAD_ASYNC_TO_LDS_B128 into ping-pong buffers.
#define ALD 40   // LDS leading dim (bf16 elems), %8==0 keeps 16B alignment
template<int OUTBF, int DOGELU, int HASBIAS, int RESADD>
__global__ __launch_bounds__(256) void gemm_bf16_kernel(
    const __bf16* __restrict__ X, const __bf16* __restrict__ Wt,
    const float* __restrict__ bias, float* __restrict__ Yf, __bf16* __restrict__ Yb,
    int M, int N, int K) {
  __shared__ __bf16 As[2][64 * ALD];
  __shared__ __bf16 Bs[2][128 * ALD];
  int bn = blockIdx.x * 128, bm = blockIdx.y * 64;
  int tid = threadIdx.x, lane = tid & 31, w = tid >> 5;
  int m0 = (w & 1) * 32, n0 = (w >> 1) * 32;
  v8f acc[2][2];
#pragma unroll
  for (int i = 0; i < 2; i++)
#pragma unroll
    for (int j = 0; j < 2; j++) acc[i][j] = (v8f){0,0,0,0,0,0,0,0};

  // per-thread async-copy assignments (16B chunks): 1 for A (64x32), 2 for B (128x32)
  int cr = tid >> 2, cc = (tid & 3) * 8;
  unsigned int voffA  = (unsigned int)((((size_t)(bm + cr)) * K + cc) * 2);
  unsigned int voffB0 = (unsigned int)((((size_t)(bn + cr)) * K + cc) * 2);
  unsigned int voffB1 = (unsigned int)((((size_t)(bn + cr + 64)) * K + cc) * 2);
  unsigned int ldsA[2], ldsB0[2], ldsB1[2];
#pragma unroll
  for (int bf = 0; bf < 2; bf++) {
    ldsA[bf]  = (unsigned int)(uintptr_t)&As[bf][cr * ALD + cc];
    ldsB0[bf] = (unsigned int)(uintptr_t)&Bs[bf][cr * ALD + cc];
    ldsB1[bf] = (unsigned int)(uintptr_t)&Bs[bf][(cr + 64) * ALD + cc];
  }
  auto issue = [&](int bf, int k0) {
    unsigned int ko = (unsigned int)(k0 * 2);
    async_copy16(ldsA[bf],  X,  voffA  + ko);
    async_copy16(ldsB0[bf], Wt, voffB0 + ko);
    async_copy16(ldsB1[bf], Wt, voffB1 + ko);
  };

  int nk = K >> 5;
  issue(0, 0);
  for (int kt = 0; kt < nk; kt++) {
    int cur = kt & 1;
    wait_async0();          // my async writes for stage kt landed in LDS
    __syncthreads();        // everyone's landed; everyone done reading buffer cur^1
    if (kt + 1 < nk) issue(cur ^ 1, (kt + 1) << 5);  // overlap next transfer with WMMA
    const __bf16* Ab = &As[cur][0];
    const __bf16* Bb = &Bs[cur][0];
    v16bf a0 = load_fragA(&Ab[m0 * ALD], ALD, lane);
    v16bf a1 = load_fragA(&Ab[(m0 + 16) * ALD], ALD, lane);
    v16bf b0 = load_fragB(&Bb[n0 * ALD], ALD, lane);
    v16bf b1 = load_fragB(&Bb[(n0 + 16) * ALD], ALD, lane);
    acc[0][0] = wmma_bf16(a0, b0, acc[0][0]);
    acc[0][1] = wmma_bf16(a0, b1, acc[0][1]);
    acc[1][0] = wmma_bf16(a1, b0, acc[1][0]);
    acc[1][1] = wmma_bf16(a1, b1, acc[1][1]);
  }
  int rl = lane & 15, hl = lane >> 4;
#pragma unroll
  for (int mf = 0; mf < 2; mf++)
#pragma unroll
    for (int nf = 0; nf < 2; nf++)
#pragma unroll
      for (int j = 0; j < 8; j++) {
        int row = bm + m0 + mf * 16 + j + 8 * hl;
        int col = bn + n0 + nf * 16 + rl;
        float v = acc[mf][nf][j];
        if (HASBIAS) v += bias[col];
        if (DOGELU) v = gelu_exact(v);
        size_t o = (size_t)row * N + col;
        if (RESADD)      Yf[o] += v;
        else if (OUTBF)  Yb[o] = (__bf16)v;
        else             Yf[o] = v;
      }
}

// ---------------- flash attention ----------------
// 1 block = (b, h, 64-row q tile); 128 threads = 4 waves, each wave owns 16 q rows.
#define SLD 72   // LDS leading dim (bf16), %8==0
__global__ __launch_bounds__(128) void flash_attn_kernel(
    const __bf16* __restrict__ Q, const __bf16* __restrict__ Km,
    const __bf16* __restrict__ Vm, __bf16* __restrict__ Ob) {
  __shared__ __bf16 Qs[64 * SLD], Ks[64 * SLD], Vt[64 * SLD], Ps[64 * SLD];
  int bid = blockIdx.x;
  int qt = bid & 15;
  int h  = (bid >> 4) % HH;
  int b  = bid / (16 * HH);
  int qbase = qt * 64;
  const __bf16* Qp = Q  + (((size_t)(b * HH + h)) * TT + qbase) * HD;
  const __bf16* Kp = Km + (((size_t)(b * HH + h)) * TT) * HD;
  const __bf16* Vp = Vm + (((size_t)(b * HH + h)) * TT) * HD;
  int tid = threadIdx.x, lane = tid & 31, w = tid >> 5;
  int m0 = w * 16, rl = lane & 15, hl = lane >> 4;

  // load Q tile (64x64)
#pragma unroll
  for (int i = 0; i < 4; i++) {
    int ch = i * 128 + tid, r = ch >> 3, cc = (ch & 7) * 8;
    *(uint4*)&Qs[r * SLD + cc] = *(const uint4*)&Qp[(size_t)r * HD + cc];
  }
  __syncthreads();
  v16bf aq0 = load_fragA(&Qs[m0 * SLD], SLD, lane);
  v16bf aq1 = load_fragA(&Qs[m0 * SLD + 32], SLD, lane);

  float rowm[8], rowl[8];
  v8f o[4];
#pragma unroll
  for (int j = 0; j < 8; j++) { rowm[j] = NEG_BIG; rowl[j] = 0.0f; }
#pragma unroll
  for (int hf = 0; hf < 4; hf++) o[hf] = (v8f){0,0,0,0,0,0,0,0};

  for (int j0 = 0; j0 <= qbase; j0 += 64) {
    __syncthreads();
#pragma unroll
    for (int i = 0; i < 4; i++) {                 // K tile + V tile (transposed)
      int ch = i * 128 + tid, r = ch >> 3, cc = (ch & 7) * 8;
      *(uint4*)&Ks[r * SLD + cc] = *(const uint4*)&Kp[(size_t)(j0 + r) * HD + cc];
      uint4 vv = *(const uint4*)&Vp[(size_t)(j0 + r) * HD + cc];
      const __bf16* ve = (const __bf16*)&vv;
#pragma unroll
      for (int e = 0; e < 8; e++) Vt[(cc + e) * SLD + r] = ve[e];
    }
    __syncthreads();

    // S = Q @ K^T  (16 x 64 per wave)
    v8f s[4];
#pragma unroll
    for (int nf = 0; nf < 4; nf++) {
      v8f a = (v8f){0,0,0,0,0,0,0,0};
      v16bf bk0 = load_fragB(&Ks[(nf * 16) * SLD], SLD, lane);
      v16bf bk1 = load_fragB(&Ks[(nf * 16) * SLD + 32], SLD, lane);
      a = wmma_bf16(aq0, bk0, a);
      a = wmma_bf16(aq1, bk1, a);
      s[nf] = a;
    }
    // scale + causal mask (only diagonal tile needs masking)
    bool diag = (j0 == qbase);
#pragma unroll
    for (int nf = 0; nf < 4; nf++)
#pragma unroll
      for (int j = 0; j < 8; j++) {
        float val = s[nf][j] * 0.125f;            // 1/sqrt(64)
        if (diag) {
          int col = j0 + nf * 16 + rl;
          int row = qbase + m0 + j + 8 * hl;
          if (col > row) val = NEG_BIG;
        }
        s[nf][j] = val;
      }
    // online softmax per row (row lives in a 16-lane half-wave)
    float mnew[8], alpha[8];
#pragma unroll
    for (int j = 0; j < 8; j++) {
      float mx = fmaxf(fmaxf(s[0][j], s[1][j]), fmaxf(s[2][j], s[3][j]));
#pragma unroll
      for (int st = 1; st < 16; st <<= 1) mx = fmaxf(mx, __shfl_xor(mx, st, 32));
      mnew[j] = fmaxf(rowm[j], mx);
      alpha[j] = __expf(rowm[j] - mnew[j]);
      rowm[j] = mnew[j];
    }
#pragma unroll
    for (int j = 0; j < 8; j++) {
      float rs = 0.0f;
#pragma unroll
      for (int nf = 0; nf < 4; nf++) {
        float p = __expf(s[nf][j] - mnew[j]);
        s[nf][j] = p;
        rs += p;
      }
#pragma unroll
      for (int st = 1; st < 16; st <<= 1) rs += __shfl_xor(rs, st, 32);
      rowl[j] = rowl[j] * alpha[j] + rs;
    }
    // stage P (this wave's 16 rows only)
#pragma unroll
    for (int nf = 0; nf < 4; nf++)
#pragma unroll
      for (int j = 0; j < 8; j++)
        Ps[(m0 + j + 8 * hl) * SLD + nf * 16 + rl] = (__bf16)s[nf][j];
    // rescale running O
#pragma unroll
    for (int hf = 0; hf < 4; hf++)
#pragma unroll
      for (int j = 0; j < 8; j++) o[hf][j] *= alpha[j];
    // O += P @ V
    v16bf pa0 = load_fragA(&Ps[m0 * SLD], SLD, lane);
    v16bf pa1 = load_fragA(&Ps[m0 * SLD + 32], SLD, lane);
#pragma unroll
    for (int hf = 0; hf < 4; hf++) {
      v16bf bv0 = load_fragB(&Vt[(hf * 16) * SLD], SLD, lane);
      v16bf bv1 = load_fragB(&Vt[(hf * 16) * SLD + 32], SLD, lane);
      o[hf] = wmma_bf16(pa0, bv0, o[hf]);
      o[hf] = wmma_bf16(pa1, bv1, o[hf]);
    }
  }
  // epilogue: O / l  ->  attn bf16 [B*T, C] (head-interleaved)
  float inv[8];
#pragma unroll
  for (int j = 0; j < 8; j++) inv[j] = 1.0f / rowl[j];
#pragma unroll
  for (int hf = 0; hf < 4; hf++)
#pragma unroll
    for (int j = 0; j < 8; j++) {
      int tok = qbase + m0 + j + 8 * hl;
      size_t off = ((size_t)(b * TT + tok)) * CC + h * HD + hf * 16 + rl;
      Ob[off] = (__bf16)(o[hf][j] * inv[j]);
    }
}

// ---------------- host side ----------------
static void run_gemm(hipStream_t s, const __bf16* X, const __bf16* Wt, const float* bias,
                     float* Yf, __bf16* Yb, int M, int N, int K, int mode) {
  dim3 g(N / 128, M / 64), blk(256);
  switch (mode) {
    case 0: gemm_bf16_kernel<1,0,0,0><<<g,blk,0,s>>>(X,Wt,nullptr,nullptr,Yb,M,N,K); break; // bf16 out
    case 1: gemm_bf16_kernel<0,0,0,1><<<g,blk,0,s>>>(X,Wt,nullptr,Yf,nullptr,M,N,K); break; // f32 +=
    case 2: gemm_bf16_kernel<1,1,1,0><<<g,blk,0,s>>>(X,Wt,bias,nullptr,Yb,M,N,K);   break; // bias+gelu -> bf16
    case 3: gemm_bf16_kernel<0,0,1,1><<<g,blk,0,s>>>(X,Wt,bias,Yf,nullptr,M,N,K);   break; // bias, f32 +=
    case 4: gemm_bf16_kernel<0,0,0,0><<<g,blk,0,s>>>(X,Wt,nullptr,Yf,nullptr,M,N,K); break; // f32 out
  }
}

extern "C" void kernel_launch(void* const* d_in, const int* in_sizes, int n_in,
                              void* d_out, int out_size, void* d_ws, size_t ws_size,
                              hipStream_t stream) {
  const int*   idx     = (const int*)  d_in[0];
  const float* tok_emb = (const float*)d_in[1];
  const float* pos_emb = (const float*)d_in[2];
  const float* Wqkv    = (const float*)d_in[3];
  const float* Wproj   = (const float*)d_in[4];
  const float* W1      = (const float*)d_in[5];
  const float* b1      = (const float*)d_in[6];
  const float* W2      = (const float*)d_in[7];
  const float* b2      = (const float*)d_in[8];
  const float* ln1_g   = (const float*)d_in[9];
  const float* ln1_b   = (const float*)d_in[10];
  const float* ln2_g   = (const float*)d_in[11];
  const float* ln2_b   = (const float*)d_in[12];
  const float* lnf_g   = (const float*)d_in[13];
  const float* lnf_b   = (const float*)d_in[14];
  float* out = (float*)d_out;

  // workspace carve-up (256B aligned)
  char* ws = (char*)d_ws;
  size_t off = 0;
  auto carve = [&](size_t bytes) { size_t o = off; off = (off + bytes + 255) & ~(size_t)255; return o; };
  float*  x      = (float*) (ws + carve((size_t)MTOK * CC * 4));
  __bf16* xnb    = (__bf16*)(ws + carve((size_t)MTOK * CC * 2));
  __bf16* qkvb   = (__bf16*)(ws + carve((size_t)MTOK * 3 * CC * 2));
  __bf16* qb     = (__bf16*)(ws + carve((size_t)MTOK * CC * 2));
  __bf16* kb     = (__bf16*)(ws + carve((size_t)MTOK * CC * 2));
  __bf16* vb     = (__bf16*)(ws + carve((size_t)MTOK * CC * 2));
  __bf16* attnb  = (__bf16*)(ws + carve((size_t)MTOK * CC * 2));
  __bf16* hb     = (__bf16*)(ws + carve((size_t)MTOK * DFF * 2));
  __bf16* WqkvT  = (__bf16*)(ws + carve((size_t)LAYERS * 3 * CC * CC * 2));
  __bf16* WprojT = (__bf16*)(ws + carve((size_t)LAYERS * CC * CC * 2));
  __bf16* W1T    = (__bf16*)(ws + carve((size_t)LAYERS * CC * DFF * 2));
  __bf16* W2T    = (__bf16*)(ws + carve((size_t)LAYERS * CC * DFF * 2));
  __bf16* tembb  = (__bf16*)(ws + carve((size_t)VV * CC * 2));
  (void)ws_size; (void)n_in; (void)in_sizes; (void)out_size;

  // weight convert/transpose (deterministic every call)
  for (int l = 0; l < LAYERS; l++) {
    { int K = CC, N = 3 * CC, n = K * N;
      wt_transpose_kernel<<<(n + 255) / 256, 256, 0, stream>>>(Wqkv + (size_t)l * n, WqkvT + (size_t)l * n, K, N); }
    { int K = CC, N = CC, n = K * N;
      wt_transpose_kernel<<<(n + 255) / 256, 256, 0, stream>>>(Wproj + (size_t)l * n, WprojT + (size_t)l * n, K, N); }
    { int K = CC, N = DFF, n = K * N;
      wt_transpose_kernel<<<(n + 255) / 256, 256, 0, stream>>>(W1 + (size_t)l * n, W1T + (size_t)l * n, K, N); }
    { int K = DFF, N = CC, n = K * N;
      wt_transpose_kernel<<<(n + 255) / 256, 256, 0, stream>>>(W2 + (size_t)l * n, W2T + (size_t)l * n, K, N); }
  }
  convert_kernel<<<(VV * CC + 255) / 256, 256, 0, stream>>>(tok_emb, tembb, VV * CC);

  // embedding
  embed_kernel<<<(MTOK * CC + 255) / 256, 256, 0, stream>>>(idx, tok_emb, pos_emb, x);

  for (int l = 0; l < LAYERS; l++) {
    // --- attention ---
    layernorm_kernel<<<MTOK, 256, 0, stream>>>(x, ln1_g + (size_t)l * CC, ln1_b + (size_t)l * CC, xnb);
    run_gemm(stream, xnb, WqkvT + (size_t)l * 3 * CC * CC, nullptr, nullptr, qkvb,
             MTOK, 3 * CC, CC, 0);
    qkv_split_kernel<<<(MTOK * CC + 255) / 256, 256, 0, stream>>>(qkvb, qb, kb, vb);
    flash_attn_kernel<<<BB * HH * (TT / 64), 128, 0, stream>>>(qb, kb, vb, attnb);
    run_gemm(stream, attnb, WprojT + (size_t)l * CC * CC, nullptr, x, nullptr,
             MTOK, CC, CC, 1);
    // --- MLP ---
    layernorm_kernel<<<MTOK, 256, 0, stream>>>(x, ln2_g + (size_t)l * CC, ln2_b + (size_t)l * CC, xnb);
    run_gemm(stream, xnb, W1T + (size_t)l * CC * DFF, b1 + (size_t)l * DFF, nullptr, hb,
             MTOK, DFF, CC, 2);
    run_gemm(stream, hb, W2T + (size_t)l * CC * DFF, b2 + (size_t)l * CC, x, nullptr,
             MTOK, CC, DFF, 3);
  }

  // final layernorm + tied-embedding logits: [8192,768] x [512,768]^T -> [8192,512]
  layernorm_kernel<<<MTOK, 256, 0, stream>>>(x, lnf_g, lnf_b, xnb);
  run_gemm(stream, xnb, tembb, nullptr, out, nullptr, MTOK, VV, CC, 4);
}
